// SlotAttention_81398220193858
// MI455X (gfx1250) — matrix operand order repaired and checked by reading, not compiled, not run on hardware
//
#include <hip/hip_runtime.h>

typedef __attribute__((ext_vector_type(16))) _Float16 v16h;
typedef __attribute__((ext_vector_type(8)))  float    v8f;
typedef _Float16 half_t;

#define BB 16
#define NN 16384
#define KS 16      // slots
#define DD 256

union frag16 { v16h h; uint4 q[2]; };

// A fragment (16x32 f16, MxK): lane: m = L&15, khalf = L>>4;
// chunks at K offsets khalf*8..+7 and 16+khalf*8..+7 (two 16B loads)
__device__ __forceinline__ v16h load_frag_a(const half_t* base, int ld, int lane) {
    int m = lane & 15, kh = lane >> 4;
    const half_t* p = base + (size_t)m * ld + kh * 8;
    frag16 f;
    f.q[0] = *(const uint4*)(p);
    f.q[1] = *(const uint4*)(p + 16);
    return f.h;
}

// B fragment (32x16 f16, KxN), stored as N rows of K-contiguous data (i.e. B^T
// row-major): lane: n = L&15, khalf = L>>4; chunks at K offsets khalf*8..+7 and 16+khalf*8..+7
__device__ __forceinline__ v16h load_frag_b(const half_t* base, int ld, int lane) {
    int n = lane & 15, kh = lane >> 4;
    const half_t* p = base + (size_t)n * ld + kh * 8;
    frag16 f;
    f.q[0] = *(const uint4*)(p);
    f.q[1] = *(const uint4*)(p + 16);
    return f.h;
}

__device__ __forceinline__ v8f wmma16(v16h a, v16h b, v8f c) {
    return __builtin_amdgcn_wmma_f32_16x16x32_f16(false, a, false, b, (short)0, c, false, false);
}

// ---------------- f32 -> f16 convert ----------------
__global__ void cvt_f16_kernel(const float* __restrict__ src, half_t* __restrict__ dst, int n) {
    int i = blockIdx.x * blockDim.x + threadIdx.x;
    int stride = gridDim.x * blockDim.x;
    for (; i < n; i += stride) dst[i] = (half_t)src[i];
}

// ---------------- zero ----------------
__global__ void zero_kernel(float* __restrict__ p, int n) {
    int i = blockIdx.x * blockDim.x + threadIdx.x;
    int stride = gridDim.x * blockDim.x;
    for (; i < n; i += stride) p[i] = 0.0f;
}

// ---------------- LN rows of 256 -> f16 ----------------
__global__ void ln256_f16_kernel(const float* __restrict__ x, const float* __restrict__ g,
                                 const float* __restrict__ b, half_t* __restrict__ out, int rows) {
    int row = blockIdx.x;
    int t = threadIdx.x;
    __shared__ float red[256];
    float xv = x[(size_t)row * 256 + t];
    red[t] = xv; __syncthreads();
    for (int s = 128; s > 0; s >>= 1) { if (t < s) red[t] += red[t + s]; __syncthreads(); }
    float mean = red[0] * (1.0f / 256.0f);
    __syncthreads();
    float d = xv - mean;
    red[t] = d * d; __syncthreads();
    for (int s = 128; s > 0; s >>= 1) { if (t < s) red[t] += red[t + s]; __syncthreads(); }
    float rstd = rsqrtf(red[0] * (1.0f / 256.0f) + 1e-5f);
    out[(size_t)row * 256 + t] = (half_t)(d * rstd * g[t] + b[t]);
}

// ---------------- fused LN(features) + k/v projection (WMMA) ----------------
// grid: B*N/16 blocks, 256 threads (8 waves). Each block: 16 tokens.
// kh[b][n][d] row-major f16 ; vT[b][d][n] d-major f16.
__global__ void ln_kv_kernel(const float* __restrict__ feat, const float* __restrict__ g,
                             const float* __restrict__ b, const half_t* __restrict__ WkH,
                             const half_t* __restrict__ WvH, half_t* __restrict__ kh,
                             half_t* __restrict__ vT) {
    __shared__ half_t sA[16 * 264];   // ld=264 (mult of 8 -> 16B aligned chunks)
    int row0 = blockIdx.x * 16;       // global token row (b folded into rows)
    int tid = threadIdx.x;
    // --- LN: thread t -> row r = t>>4, cols c0 + i*16 ---
    int r = tid >> 4, c0 = tid & 15;
    const float* x = feat + (size_t)(row0 + r) * 256;
    float s = 0.0f, s2 = 0.0f;
    float vals[16];
#pragma unroll
    for (int i = 0; i < 16; ++i) { float xv = x[c0 + i * 16]; vals[i] = xv; s += xv; s2 += xv * xv; }
#pragma unroll
    for (int m = 8; m >= 1; m >>= 1) { s += __shfl_xor(s, m, 32); s2 += __shfl_xor(s2, m, 32); }
    float mean = s * (1.0f / 256.0f);
    float var  = s2 * (1.0f / 256.0f) - mean * mean;
    float rstd = rsqrtf(var + 1e-5f);
#pragma unroll
    for (int i = 0; i < 16; ++i) {
        int col = c0 + i * 16;
        sA[r * 264 + col] = (half_t)((vals[i] - mean) * rstd * g[col] + b[col]);
    }
    __syncthreads();
    // --- WMMA: 8 waves x 4 combos = 32 = {k,v} x 16 d-tiles ---
    int wave = tid >> 5, lane = tid & 31;
    // A-fragments depend only on lane: load once from LDS, reuse for 4 tiles
    v16h afr[8];
#pragma unroll
    for (int kk = 0; kk < 8; ++kk) afr[kk] = load_frag_a(sA + kk * 32, 264, lane);
#pragma unroll
    for (int it = 0; it < 4; ++it) {
        int combo = wave * 4 + it;
        int mat = combo >> 4;          // 0 = k, 1 = v
        int nt  = combo & 15;          // d tile
        const half_t* W = mat ? WvH : WkH;
        v8f c = {};
#pragma unroll
        for (int kk = 0; kk < 8; ++kk) {
            v16h bf = load_frag_b(W + (size_t)nt * 16 * 256 + kk * 32, 256, lane);
            c = wmma16(afr[kk], bf, c);
        }
        int nidx = lane & 15, rj0 = (lane >> 4) * 8;
        if (mat == 0) {
#pragma unroll
            for (int j = 0; j < 8; ++j) {
                int rr = row0 + rj0 + j;
                kh[(size_t)rr * 256 + nt * 16 + nidx] = (half_t)c[j];
            }
        } else {
            int bidx = row0 >> 14;            // row0 / N
            int btok = row0 & (NN - 1);
            half_t tmp[8];
#pragma unroll
            for (int j = 0; j < 8; ++j) tmp[j] = (half_t)c[j];
            size_t o = (size_t)bidx * DD * NN + (size_t)(nt * 16 + nidx) * NN + btok + rj0;
            *(uint4*)(vT + o) = *(const uint4*)tmp;   // 8 consecutive tokens, 16B
        }
    }
}

// ---------------- q = LN(slots) @ Wq^T (WMMA), [256,256] ----------------
__global__ void q_gemm_kernel(const half_t* __restrict__ A, const half_t* __restrict__ Bw,
                              half_t* __restrict__ out) {
    int mt = blockIdx.x, nt = blockIdx.y;
    int lane = threadIdx.x;
    v8f c = {};
#pragma unroll
    for (int kk = 0; kk < 8; ++kk) {
        v16h a  = load_frag_a(A  + (size_t)mt * 16 * 256 + kk * 32, 256, lane);
        v16h bf = load_frag_b(Bw + (size_t)nt * 16 * 256 + kk * 32, 256, lane);
        c = wmma16(a, bf, c);
    }
    int nc = lane & 15, rj0 = (lane >> 4) * 8;
#pragma unroll
    for (int j = 0; j < 8; ++j)
        out[(size_t)(mt * 16 + rj0 + j) * 256 + nt * 16 + nc] = (half_t)c[j];
}

// ---------------- logits + softmax-over-slots + attn (WMMA) ----------------
// grid (N/128, B), 32 threads. Writes attn f16 [b][slot][n], atomicAdds rowsum,
// optionally writes f32 attn to d_out on the final iteration.
__global__ void attn_kernel(const half_t* __restrict__ qh, const half_t* __restrict__ kh,
                            half_t* __restrict__ attnH, float* __restrict__ rowsum,
                            float* __restrict__ attnOut, int writeF32) {
    int b = blockIdx.y;
    int n0 = blockIdx.x * 128;
    int lane = threadIdx.x;
    __shared__ float sAttn[16][129];
    const half_t* qbase = qh + (size_t)b * 16 * 256;
    const half_t* kbase = kh + ((size_t)b * NN + n0) * 256;
    int bn = lane & 15, bkh = lane >> 4;
    v16h afr[8];
#pragma unroll
    for (int kk = 0; kk < 8; ++kk) afr[kk] = load_frag_a(qbase + kk * 32, 256, lane);
    const float scale = 0.0625f;   // D^-0.5
    for (int st = 0; st < 8; ++st) {
        // prefetch next sub-tile's B rows (single-wave block: no other wave hides latency)
        if (st + 1 < 8)
            __builtin_prefetch(kbase + (size_t)(st + 1) * 16 * 256 + (size_t)bn * 256 + bkh * 16, 0, 1);
        v8f c = {};
#pragma unroll
        for (int kk = 0; kk < 8; ++kk) {
            v16h bf = load_frag_b(kbase + (size_t)st * 16 * 256 + kk * 32, 256, lane);
            c = wmma16(afr[kk], bf, c);
        }
        // softmax over 16 slots: 8 in-lane + partner lane (L ^ 16)
        float mx = -1e30f;
#pragma unroll
        for (int j = 0; j < 8; ++j) { c[j] *= scale; mx = fmaxf(mx, c[j]); }
        mx = fmaxf(mx, __shfl_xor(mx, 16, 32));
        float e[8]; float ssum = 0.0f;
#pragma unroll
        for (int j = 0; j < 8; ++j) { e[j] = __expf(c[j] - mx); ssum += e[j]; }
        ssum += __shfl_xor(ssum, 16, 32);
        float inv = 1.0f / ssum;
        int ncol = lane & 15, rj0 = (lane >> 4) * 8;
#pragma unroll
        for (int j = 0; j < 8; ++j) sAttn[rj0 + j][st * 16 + ncol] = e[j] * inv;
    }
    __syncthreads();
    // per-slot partial sums over the 128 tokens
    {
        int slot = lane >> 1, hh = lane & 1;
        float ps = 0.0f;
        for (int t = 0; t < 64; ++t) ps += sAttn[slot][hh * 64 + t];
        ps += __shfl_xor(ps, 1, 32);
        if (hh == 0) unsafeAtomicAdd(rowsum + b * 16 + slot, ps);
    }
    // write attn
    for (int idx = lane; idx < 16 * 128; idx += 32) {
        int slot = idx >> 7, t = idx & 127;
        float a = sAttn[slot][t];
        size_t o = ((size_t)(b * 16 + slot) << 14) + n0 + t;
        attnH[o] = (half_t)a;
        if (writeF32) attnOut[o] = a;
    }
}

// ---------------- updates = attn @ v (WMMA, split-K, f32 atomics) ----------------
// grid (16 dtiles, 8 ksplit, B), 32 threads
__global__ void updates_kernel(const half_t* __restrict__ attnH, const half_t* __restrict__ vT,
                               float* __restrict__ updates) {
    int dt = blockIdx.x, ks = blockIdx.y, b = blockIdx.z;
    int lane = threadIdx.x;
    const int nPer = NN / 8;   // 2048
    const int nIt = nPer / 32; // 64
    int n0 = ks * nPer;
    const half_t* abase = attnH + (size_t)b * 16 * NN + n0;
    const half_t* vbase = vT + (size_t)b * DD * NN + (size_t)dt * 16 * NN + n0;
    int fm = lane & 15, fkh = lane >> 4;
    const half_t* apre = abase + (size_t)fm * NN + fkh * 8;
    const half_t* vpre = vbase + (size_t)fm * NN + fkh * 16;
    v8f c = {};
    for (int kk = 0; kk < nIt; ++kk) {
        if (kk + 8 < nIt) {   // stream-ahead prefetch (512B / 4 lines ahead per lane-row)
            __builtin_prefetch(apre + (kk + 8) * 32, 0, 1);
            __builtin_prefetch(vpre + (kk + 8) * 32, 0, 1);
        }
        v16h a  = load_frag_a(abase + kk * 32, NN, lane);
        v16h bf = load_frag_b(vbase + kk * 32, NN, lane);
        c = wmma16(a, bf, c);
    }
    int d = dt * 16 + (lane & 15), sj0 = (lane >> 4) * 8;
#pragma unroll
    for (int j = 0; j < 8; ++j)
        unsafeAtomicAdd(updates + (size_t)(b * 16 + sj0 + j) * 256 + d, c[j]);
}

// ---------------- GRU cell + residual MLP, one block per (b,slot) row ----------------
__global__ void gru_mlp_kernel(float* __restrict__ slots, const float* __restrict__ updates,
                               const float* __restrict__ rowsum,
                               const float* __restrict__ w_ih, const float* __restrict__ w_hh,
                               const float* __restrict__ b_ih, const float* __restrict__ b_hh,
                               const float* __restrict__ g_m, const float* __restrict__ b_m,
                               const float* __restrict__ W1, const float* __restrict__ b1,
                               const float* __restrict__ W2, const float* __restrict__ b2) {
    int row = blockIdx.x;    // b*16 + slot
    int t = threadIdx.x;     // 256 threads
    __shared__ float sx[256], sh[256], sln[256], shid[1024], red[256];
    float inv = 1.0f / (rowsum[row] + 1.0f);
    sx[t] = updates[(size_t)row * 256 + t] * inv;
    float h = slots[(size_t)row * 256 + t];
    sh[t] = h;
    __syncthreads();
    float gi[3], gh[3];
#pragma unroll
    for (int g = 0; g < 3; ++g) {
        int o = g * 256 + t;
        float ai = b_ih[o], ah = b_hh[o];
        const float* wi = w_ih + (size_t)o * 256;
        const float* wh = w_hh + (size_t)o * 256;
        for (int d = 0; d < 256; ++d) { ai += wi[d] * sx[d]; ah += wh[d] * sh[d]; }
        gi[g] = ai; gh[g] = ah;
    }
    float r = 1.0f / (1.0f + __expf(-(gi[0] + gh[0])));
    float z = 1.0f / (1.0f + __expf(-(gi[1] + gh[1])));
    float n = tanhf(gi[2] + r * gh[2]);
    float hnew = (1.0f - z) * n + z * h;
    // LayerNorm(hnew)
    red[t] = hnew; __syncthreads();
    for (int s = 128; s > 0; s >>= 1) { if (t < s) red[t] += red[t + s]; __syncthreads(); }
    float mean = red[0] * (1.0f / 256.0f);
    __syncthreads();
    float dd = hnew - mean;
    red[t] = dd * dd; __syncthreads();
    for (int s = 128; s > 0; s >>= 1) { if (t < s) red[t] += red[t + s]; __syncthreads(); }
    float rstd = rsqrtf(red[0] * (1.0f / 256.0f) + 1e-5f);
    sln[t] = dd * rstd * g_m[t] + b_m[t];
    __syncthreads();
    // hidden = relu(ln @ W1^T + b1)
#pragma unroll
    for (int i = 0; i < 4; ++i) {
        int j = i * 256 + t;
        float acc = b1[j];
        const float* w = W1 + (size_t)j * 256;
        for (int d = 0; d < 256; ++d) acc += w[d] * sln[d];
        shid[j] = fmaxf(acc, 0.0f);
    }
    __syncthreads();
    float acc = b2[t];
    const float* w2 = W2 + (size_t)t * 1024;
    for (int j = 0; j < 1024; ++j) acc += w2[j] * shid[j];
    slots[(size_t)row * 256 + t] = hnew + acc;
}

extern "C" void kernel_launch(void* const* d_in, const int* in_sizes, int n_in,
                              void* d_out, int out_size, void* d_ws, size_t ws_size,
                              hipStream_t stream) {
    (void)in_sizes; (void)n_in; (void)out_size; (void)ws_size;
    const float* features   = (const float*)d_in[0];
    const float* slots_init = (const float*)d_in[1];
    const float* g_f = (const float*)d_in[2];
    const float* b_f = (const float*)d_in[3];
    const float* g_s = (const float*)d_in[4];
    const float* b_s = (const float*)d_in[5];
    const float* g_m = (const float*)d_in[6];
    const float* b_m = (const float*)d_in[7];
    const float* Wq  = (const float*)d_in[8];
    const float* Wk  = (const float*)d_in[9];
    const float* Wv  = (const float*)d_in[10];
    const float* w_ih = (const float*)d_in[11];
    const float* w_hh = (const float*)d_in[12];
    const float* b_ih = (const float*)d_in[13];
    const float* b_hh = (const float*)d_in[14];
    const float* W1  = (const float*)d_in[15];
    const float* b1  = (const float*)d_in[16];
    const float* W2  = (const float*)d_in[17];
    const float* b2  = (const float*)d_in[18];

    char* ws = (char*)d_ws;
    size_t off = 0;
    auto carve = [&](size_t bytes) { size_t o = (off + 255) & ~(size_t)255; off = o + bytes; return ws + o; };

    half_t* kh      = (half_t*)carve((size_t)BB * NN * DD * 2);
    half_t* vT      = (half_t*)carve((size_t)BB * NN * DD * 2);
    half_t* attnH   = (half_t*)carve((size_t)BB * KS * NN * 2);
    half_t* qh      = (half_t*)carve((size_t)BB * KS * DD * 2);
    half_t* lnslots = (half_t*)carve((size_t)BB * KS * DD * 2);
    half_t* WqH     = (half_t*)carve((size_t)DD * DD * 2);
    half_t* WkH     = (half_t*)carve((size_t)DD * DD * 2);
    half_t* WvH     = (half_t*)carve((size_t)DD * DD * 2);
    float*  slots   = (float*)carve((size_t)BB * KS * DD * 4);
    float*  updates = (float*)carve((size_t)BB * KS * DD * 4);
    float*  rowsum  = (float*)carve((size_t)BB * KS * 4);

    // weights -> f16
    cvt_f16_kernel<<<64, 256, 0, stream>>>(Wq, WqH, DD * DD);
    cvt_f16_kernel<<<64, 256, 0, stream>>>(Wk, WkH, DD * DD);
    cvt_f16_kernel<<<64, 256, 0, stream>>>(Wv, WvH, DD * DD);

    // fused LN(features) + k/v projections
    ln_kv_kernel<<<(BB * NN) / 16, 256, 0, stream>>>(features, g_f, b_f, WkH, WvH, kh, vT);

    // slots <- slots_init
    hipMemcpyAsync(slots, slots_init, (size_t)BB * KS * DD * 4, hipMemcpyDeviceToDevice, stream);

    float* attnOut = (float*)d_out + (size_t)BB * KS * DD;
    for (int iter = 0; iter < 3; ++iter) {
        int last = (iter == 2);
        ln256_f16_kernel<<<BB * KS, 256, 0, stream>>>(slots, g_s, b_s, lnslots, BB * KS);
        q_gemm_kernel<<<dim3(16, 16), 32, 0, stream>>>(lnslots, WqH, qh);
        zero_kernel<<<64, 256, 0, stream>>>(updates, BB * KS * DD);
        zero_kernel<<<1, 256, 0, stream>>>(rowsum, BB * KS);
        attn_kernel<<<dim3(NN / 128, BB), 32, 0, stream>>>(qh, kh, attnH, rowsum, attnOut, last);
        updates_kernel<<<dim3(16, 8, BB), 32, 0, stream>>>(attnH, vT, updates);
        gru_mlp_kernel<<<BB * KS, 256, 0, stream>>>(slots, updates, rowsum,
                                                    w_ih, w_hh, b_ih, b_hh, g_m, b_m,
                                                    W1, b1, W2, b2);
    }
    hipMemcpyAsync(d_out, slots, (size_t)BB * KS * DD * 4, hipMemcpyDeviceToDevice, stream);
}
